// TransformerImplicit_45277545235066
// MI455X (gfx1250) — compile-verified, matrix operand order checked by reading, not run on hardware
//
#include <hip/hip_runtime.h>
#include <hip/hip_bf16.h>

// ---------------------------------------------------------------------------
// Model dims (compile-time constants from the reference)
// ---------------------------------------------------------------------------
#define BSZ   16
#define CLEN  1024
#define QLEN  128
#define SLEN  1152          // CLEN + QLEN
#define DMODEL 512
#define NHEAD 8
#define HDIM  64
#define HIDDEN 2048
#define NLAYER 6
#define MTOK  (BSZ * SLEN)  // 18432 tokens

typedef __attribute__((ext_vector_type(16))) __bf16 v16bf;
typedef __attribute__((ext_vector_type(8)))  __bf16 v8bf;
typedef __attribute__((ext_vector_type(4)))  __bf16 v4bf;
typedef __attribute__((ext_vector_type(8)))  float  v8f;

__device__ __forceinline__ v8f wmma_bf16(v16bf a, v16bf b, v8f c) {
  // D = A(16x32) * B(32x16) + C, f32 accumulate
  return __builtin_amdgcn_wmma_f32_16x16x32_bf16(false, a, false, b, (short)0, c,
                                                 false, false);
}

// Fragment element e -> K = (e>>3)*16 + 8*half + (e&7). Caller pre-offsets the
// pointer by (row*ld + kbase + 8*half), so the fragment is p[0..7] ++ p[16..23].
__device__ __forceinline__ v16bf load_frag(const float* __restrict__ p) {
  v16bf r;
#pragma unroll
  for (int e = 0; e < 8; ++e) {
    r[e]     = (__bf16)p[e];
    r[e + 8] = (__bf16)p[e + 16];
  }
  return r;
}

__device__ __forceinline__ v16bf load_frag_scaled(const float* __restrict__ p,
                                                  float s) {
  v16bf r;
#pragma unroll
  for (int e = 0; e < 8; ++e) {
    r[e]     = (__bf16)(p[e] * s);
    r[e + 8] = (__bf16)(p[e + 16] * s);
  }
  return r;
}

// bf16 source (global weights or LDS tiles): two aligned 16-B vector loads.
__device__ __forceinline__ v16bf load_frag_bf(const __bf16* p) {
  v8bf lo = *(const v8bf*)p;
  v8bf hi = *(const v8bf*)(p + 16);
  return __builtin_shufflevector(lo, hi, 0, 1, 2, 3, 4, 5, 6, 7, 8, 9, 10, 11,
                                 12, 13, 14, 15);
}

// ---------------------------------------------------------------------------
// Elementwise f32 -> bf16 (weights pre-conversion; count % 1024 == 0)
// ---------------------------------------------------------------------------
__global__ __launch_bounds__(256) void cvt_bf16_kernel(
    const float* __restrict__ src, __bf16* __restrict__ dst) {
  size_t i = ((size_t)blockIdx.x * 256 + threadIdx.x) * 4;
  float4 v = *(const float4*)(src + i);
  v4bf o;
  o[0] = (__bf16)v.x;
  o[1] = (__bf16)v.y;
  o[2] = (__bf16)v.z;
  o[3] = (__bf16)v.w;
  *(v4bf*)(dst + i) = o;
}

// ---------------------------------------------------------------------------
// Embed: out[t, :] = concat(x, y) @ W_val + b_val (+ q_emb for query tokens)
// ---------------------------------------------------------------------------
__global__ __launch_bounds__(128) void embed_kernel(
    const float* __restrict__ x_c, const float* __restrict__ y_c,
    const float* __restrict__ x_q, const float* __restrict__ W_val,
    const float* __restrict__ b_val, const float* __restrict__ q_emb,
    float* __restrict__ out) {
  __shared__ float in[96];
  int t = blockIdx.x;
  int b = t / SLEN, s = t % SLEN;
  int tid = threadIdx.x;
  if (tid < 96) {
    float v;
    if (s < CLEN) {
      v = (tid < 64) ? x_c[((size_t)b * CLEN + s) * 64 + tid]
                     : y_c[((size_t)b * CLEN + s) * 32 + (tid - 64)];
    } else {
      v = (tid < 64) ? x_q[((size_t)b * QLEN + (s - CLEN)) * 64 + tid] : 0.0f;
    }
    in[tid] = v;
  }
  __syncthreads();
#pragma unroll
  for (int j = 0; j < 4; ++j) {
    int d = tid + j * 128;
    float acc = b_val[d] + (s >= CLEN ? q_emb[d] : 0.0f);
#pragma unroll 4
    for (int k = 0; k < 96; ++k) acc += in[k] * W_val[k * DMODEL + d];
    out[(size_t)t * DMODEL + d] = acc;
  }
}

// ---------------------------------------------------------------------------
// bf16 WMMA GEMM: out[M,N] = act(A[M,K] @ Wbf[N,K]^T + bias[N]) (+ res[M,N])
// Block tile 128x128; 8 waves in a 4(M) x 2(N) grid, each wave 32x64:
// 2 A-frags (f32->bf16 in-register) x 4 B-frags (bf16 direct) -> 8 WMMA/K-step.
// All grids are exact multiples -> EXEC all-ones around every WMMA.
// ---------------------------------------------------------------------------
__global__ __launch_bounds__(256) void gemm_kernel(
    const float* __restrict__ A, const __bf16* __restrict__ W,
    const float* __restrict__ bias, const float* __restrict__ res,
    float* __restrict__ out, int N, int K, int relu) {
  int w    = threadIdx.x >> 5;
  int lane = threadIdx.x & 31;
  int half = lane >> 4, nl = lane & 15;
  int wm = w >> 1, wn = w & 1;

  int r0 = blockIdx.x * 128 + wm * 32;
  int c0 = blockIdx.y * 128 + wn * 64;

  const float* a0p = A + (size_t)(r0 + nl) * K + 8 * half;
  const float* a1p = a0p + (size_t)16 * K;
  const __bf16* wp0 = W + (size_t)(c0 + nl) * K + 8 * half;
  const __bf16* wp1 = wp0 + (size_t)16 * K;
  const __bf16* wp2 = wp0 + (size_t)32 * K;
  const __bf16* wp3 = wp0 + (size_t)48 * K;

  v8f acc[2][4] = {};
  for (int kb = 0; kb < K; kb += 32) {
    v16bf a0 = load_frag(a0p + kb);
    v16bf a1 = load_frag(a1p + kb);
    v16bf b0 = load_frag_bf(wp0 + kb);
    v16bf b1 = load_frag_bf(wp1 + kb);
    v16bf b2 = load_frag_bf(wp2 + kb);
    v16bf b3 = load_frag_bf(wp3 + kb);
    acc[0][0] = wmma_bf16(a0, b0, acc[0][0]);
    acc[0][1] = wmma_bf16(a0, b1, acc[0][1]);
    acc[0][2] = wmma_bf16(a0, b2, acc[0][2]);
    acc[0][3] = wmma_bf16(a0, b3, acc[0][3]);
    acc[1][0] = wmma_bf16(a1, b0, acc[1][0]);
    acc[1][1] = wmma_bf16(a1, b1, acc[1][1]);
    acc[1][2] = wmma_bf16(a1, b2, acc[1][2]);
    acc[1][3] = wmma_bf16(a1, b3, acc[1][3]);
  }

#pragma unroll
  for (int i = 0; i < 2; ++i) {
    int rowb = r0 + i * 16 + 8 * half;  // D rows: r + 8*half
#pragma unroll
    for (int t = 0; t < 4; ++t) {
      int col = c0 + t * 16 + nl;
      float bv = bias[col];
#pragma unroll
      for (int r = 0; r < 8; ++r) {
        size_t o = (size_t)(rowb + r) * N + col;
        float v = acc[i][t][r] + bv;
        if (relu) v = fmaxf(v, 0.0f);
        if (res) v += res[o];
        out[o] = v;
      }
    }
  }
}

// ---------------------------------------------------------------------------
// Flash attention. blockIdx.y = (b,h); 8 waves = 8 query tiles of 16 sharing
// the same K/V. Per 32-key step the block cooperatively stages K (row-major)
// and V (transposed) as bf16 into padded LDS (all fragment reads = aligned
// 16-B ds loads). Wave computes S^T = K.Q^T (per-lane softmax stats), then
// O^T = V^T.P^T where the C-layout of the two S^T tiles is element-for-element
// the B-operand of the PV WMMA (no cross-lane movement). Keys limited to the
// 1024 context positions == the reference's additive -1e9 mask exactly.
// ---------------------------------------------------------------------------
#define KT_STRIDE 80  // bf16 elems per K row  (32 rows)  -> 160 B, 16-B aligned
#define VT_STRIDE 40  // bf16 elems per V^T row (64 rows) ->  80 B, 16-B aligned

__global__ __launch_bounds__(256) void flash_kernel(const float* __restrict__ qkv,
                                                    float* __restrict__ ctx) {
  __shared__ __attribute__((aligned(16))) __bf16 kt[32 * KT_STRIDE];
  __shared__ __attribute__((aligned(16))) __bf16 vt[64 * VT_STRIDE];

  int tid  = threadIdx.x;
  int lane = tid & 31;
  int half = lane >> 4, nl = lane & 15;
  int b = blockIdx.y >> 3, hh = blockIdx.y & 7;
  int qtile = blockIdx.x * 8 + (tid >> 5);  // 0..71

  size_t qtok = (size_t)b * SLEN + qtile * 16 + nl;
  const float* qrow = qkv + qtok * (3 * DMODEL) + hh * HDIM + 8 * half;
  v16bf qb0 = load_frag_scaled(qrow, 0.125f);  // 1/sqrt(HD) folded into Q
  v16bf qb1 = load_frag_scaled(qrow + 32, 0.125f);

  // cooperative staging indices: 256 threads x 8 floats = one 32x64 tile
  int ckey = tid >> 3;         // 0..31
  int chd  = (tid & 7) * 8;    // 0,8,...,56
  size_t brow = (size_t)b * SLEN;
  const float* kld = qkv + (brow + ckey) * (3 * DMODEL) + DMODEL + hh * HDIM + chd;
  const float* vld = kld + DMODEL;

  float m_run = -3.0e38f, l_run = 0.0f;
  v8f o[4] = {};

  for (int t = 0; t < CLEN / 32; ++t) {
    int k0 = t * 32;
    {  // stage K row-major and V transposed (bf16)
      const float* kp = kld + (size_t)k0 * (3 * DMODEL);
      const float* vp = vld + (size_t)k0 * (3 * DMODEL);
      v8bf kv;
#pragma unroll
      for (int j = 0; j < 8; ++j) kv[j] = (__bf16)kp[j];
      *(v8bf*)(kt + ckey * KT_STRIDE + chd) = kv;
#pragma unroll
      for (int j = 0; j < 8; ++j)
        vt[(chd + j) * VT_STRIDE + ckey] = (__bf16)vp[j];
    }
    __syncthreads();

    v8f s0 = {}, s1 = {};
    s0 = wmma_bf16(load_frag_bf(kt + nl * KT_STRIDE + 8 * half), qb0, s0);
    s0 = wmma_bf16(load_frag_bf(kt + nl * KT_STRIDE + 32 + 8 * half), qb1, s0);
    s1 = wmma_bf16(load_frag_bf(kt + (16 + nl) * KT_STRIDE + 8 * half), qb0, s1);
    s1 = wmma_bf16(load_frag_bf(kt + (16 + nl) * KT_STRIDE + 32 + 8 * half), qb1, s1);

    float mt = s0[0];
#pragma unroll
    for (int r = 0; r < 8; ++r) {
      mt = fmaxf(mt, s0[r]);
      mt = fmaxf(mt, s1[r]);
    }
    mt = fmaxf(mt, __shfl_xor(mt, 16, 32));
    float m_new = fmaxf(m_run, mt);
    float corr = __expf(m_run - m_new);

    v16bf pb;
    float psum = 0.0f;
#pragma unroll
    for (int r = 0; r < 8; ++r) {
      float p0 = __expf(s0[r] - m_new);
      float p1 = __expf(s1[r] - m_new);
      pb[r]     = (__bf16)p0;  // keys k0 + 8*half + r      (S^T tile 0)
      pb[r + 8] = (__bf16)p1;  // keys k0 + 16 + 8*half + r (S^T tile 1)
      psum += p0 + p1;
    }
    psum += __shfl_xor(psum, 16, 32);
    l_run = l_run * corr + psum;
    m_run = m_new;

#pragma unroll
    for (int tt = 0; tt < 4; ++tt)
#pragma unroll
      for (int r = 0; r < 8; ++r) o[tt][r] *= corr;

#pragma unroll
    for (int tt = 0; tt < 4; ++tt) {
      v16bf va = load_frag_bf(vt + (tt * 16 + nl) * VT_STRIDE + 8 * half);
      o[tt] = wmma_bf16(va, pb, o[tt]);
    }
    __syncthreads();
  }

  float inv_l = 1.0f / l_run;
  // O^T D-layout: lane col = q (nl), rows = hd local r + 8*half per tile.
  float* crow = ctx + qtok * DMODEL + hh * HDIM + 8 * half;
#pragma unroll
  for (int tt = 0; tt < 4; ++tt)
#pragma unroll
    for (int r = 0; r < 8; ++r) crow[tt * 16 + r] = o[tt][r] * inv_l;
}

// ---------------------------------------------------------------------------
// In-place LayerNorm over D=512 per token.
// ---------------------------------------------------------------------------
__global__ __launch_bounds__(256) void layernorm_kernel(
    float* __restrict__ x, const float* __restrict__ g,
    const float* __restrict__ bta) {
  __shared__ float s_sum[256];
  __shared__ float s_sq[256];
  int tid = threadIdx.x;
  size_t base = (size_t)blockIdx.x * DMODEL;
  float v0 = x[base + tid];
  float v1 = x[base + tid + 256];
  s_sum[tid] = v0 + v1;
  s_sq[tid] = v0 * v0 + v1 * v1;
  __syncthreads();
  for (int off = 128; off > 0; off >>= 1) {
    if (tid < off) {
      s_sum[tid] += s_sum[tid + off];
      s_sq[tid] += s_sq[tid + off];
    }
    __syncthreads();
  }
  float mu = s_sum[0] * (1.0f / DMODEL);
  float var = s_sq[0] * (1.0f / DMODEL) - mu * mu;
  float inv = rsqrtf(var + 1e-5f);
  x[base + tid]       = (v0 - mu) * inv * g[tid] + bta[tid];
  x[base + tid + 256] = (v1 - mu) * inv * g[tid + 256] + bta[tid + 256];
}

// ---------------------------------------------------------------------------
// Head: out[b,q,yd] = x[b, C+q, :] @ W_head[:, yd] + b_head[yd]
// ---------------------------------------------------------------------------
__global__ __launch_bounds__(256) void head_kernel(
    const float* __restrict__ x, const float* __restrict__ W,
    const float* __restrict__ bias, float* __restrict__ out) {
  int idx = blockIdx.x * 256 + threadIdx.x;  // 0 .. 65535
  int yd = idx & 31;
  int q = (idx >> 5) & 127;
  int b = idx >> 12;
  size_t tok = (size_t)b * SLEN + CLEN + q;
  const float* xr = x + tok * DMODEL;
  float acc = bias[yd];
#pragma unroll 4
  for (int k = 0; k < DMODEL; ++k) acc += xr[k] * W[k * 32 + yd];
  out[idx] = acc;
}

// ---------------------------------------------------------------------------
extern "C" void kernel_launch(void* const* d_in, const int* in_sizes, int n_in,
                              void* d_out, int out_size, void* d_ws,
                              size_t ws_size, hipStream_t stream) {
  (void)in_sizes; (void)n_in; (void)out_size; (void)ws_size;
  const float* x_c       = (const float*)d_in[0];
  const float* y_c       = (const float*)d_in[1];
  const float* x_q       = (const float*)d_in[2];
  const float* W_val     = (const float*)d_in[3];
  const float* b_val     = (const float*)d_in[4];
  const float* q_emb     = (const float*)d_in[5];
  const float* in_proj_w = (const float*)d_in[6];
  const float* in_proj_b = (const float*)d_in[7];
  const float* out_w     = (const float*)d_in[8];
  const float* out_b     = (const float*)d_in[9];
  const float* ln1_g     = (const float*)d_in[10];
  const float* ln1_b     = (const float*)d_in[11];
  const float* lin1_w    = (const float*)d_in[12];
  const float* lin1_b    = (const float*)d_in[13];
  const float* lin2_w    = (const float*)d_in[14];
  const float* lin2_b    = (const float*)d_in[15];
  const float* ln2_g     = (const float*)d_in[16];
  const float* ln2_b     = (const float*)d_in[17];
  const float* W_head    = (const float*)d_in[18];
  const float* b_head    = (const float*)d_in[19];

  float* xbuf = (float*)d_ws;                  // [M, D]    activations
  float* cbuf = xbuf + (size_t)MTOK * DMODEL;  // [M, D]    attn context
  float* tmp  = cbuf + (size_t)MTOK * DMODEL;  // [M, 2048] qkv / ff hidden
  __bf16* wbf = (__bf16*)(tmp + (size_t)MTOK * HIDDEN);  // bf16 weights
  const size_t n_ipw = (size_t)NLAYER * 3 * DMODEL * DMODEL;  // 4,718,592
  const size_t n_ow  = (size_t)NLAYER * DMODEL * DMODEL;      // 1,572,864
  const size_t n_l1  = (size_t)NLAYER * HIDDEN * DMODEL;      // 6,291,456
  const size_t n_l2  = (size_t)NLAYER * DMODEL * HIDDEN;      // 6,291,456
  __bf16* ipw_bf = wbf;
  __bf16* ow_bf  = ipw_bf + n_ipw;
  __bf16* l1_bf  = ow_bf + n_ow;
  __bf16* l2_bf  = l1_bf + n_l1;

  // One-shot weight conversion (L2-resident bf16 copies)
  cvt_bf16_kernel<<<n_ipw / 1024, 256, 0, stream>>>(in_proj_w, ipw_bf);
  cvt_bf16_kernel<<<n_ow / 1024, 256, 0, stream>>>(out_w, ow_bf);
  cvt_bf16_kernel<<<n_l1 / 1024, 256, 0, stream>>>(lin1_w, l1_bf);
  cvt_bf16_kernel<<<n_l2 / 1024, 256, 0, stream>>>(lin2_w, l2_bf);

  embed_kernel<<<MTOK, 128, 0, stream>>>(x_c, y_c, x_q, W_val, b_val, q_emb,
                                         xbuf);

  const int gx = MTOK / 128;  // 144
  for (int l = 0; l < NLAYER; ++l) {
    // QKV projection: [M,512] @ [1536,512]^T
    gemm_kernel<<<dim3(gx, (3 * DMODEL) / 128), 256, 0, stream>>>(
        xbuf, ipw_bf + (size_t)l * 3 * DMODEL * DMODEL,
        in_proj_b + (size_t)l * 3 * DMODEL, nullptr, tmp, 3 * DMODEL, DMODEL, 0);
    // Attention
    flash_kernel<<<dim3(SLEN / 16 / 8, BSZ * NHEAD), 256, 0, stream>>>(tmp,
                                                                       cbuf);
    // Output projection + residual (in place on xbuf)
    gemm_kernel<<<dim3(gx, DMODEL / 128), 256, 0, stream>>>(
        cbuf, ow_bf + (size_t)l * DMODEL * DMODEL, out_b + (size_t)l * DMODEL,
        xbuf, xbuf, DMODEL, DMODEL, 0);
    layernorm_kernel<<<MTOK, 256, 0, stream>>>(xbuf, ln1_g + (size_t)l * DMODEL,
                                               ln1_b + (size_t)l * DMODEL);
    // FFN up + ReLU
    gemm_kernel<<<dim3(gx, HIDDEN / 128), 256, 0, stream>>>(
        xbuf, l1_bf + (size_t)l * HIDDEN * DMODEL, lin1_b + (size_t)l * HIDDEN,
        nullptr, tmp, HIDDEN, DMODEL, 1);
    // FFN down + residual (in place)
    gemm_kernel<<<dim3(gx, DMODEL / 128), 256, 0, stream>>>(
        tmp, l2_bf + (size_t)l * DMODEL * HIDDEN, lin2_b + (size_t)l * DMODEL,
        xbuf, xbuf, DMODEL, HIDDEN, 0);
    layernorm_kernel<<<MTOK, 256, 0, stream>>>(xbuf, ln2_g + (size_t)l * DMODEL,
                                               ln2_b + (size_t)l * DMODEL);
  }

  head_kernel<<<(BSZ * QLEN * 32) / 256, 256, 0, stream>>>(xbuf, W_head, b_head,
                                                           (float*)d_out);
}
